// MultiConditionCrossAttention_24283745091773
// MI455X (gfx1250) — matrix operand order repaired and checked by reading, not compiled, not run on hardware
//
#include <hip/hip_runtime.h>
#include <hip/hip_bf16.h>
#include <math.h>

// ---------------------------------------------------------------------------
// MultiConditionCrossAttention fused kernel for gfx1250 (MI455X, wave32, WMMA)
//   x:(16,8192,512) f32 -> out:(16,8192,512) f32
//   HBM floor ~22us (512MB @ 23.3TB/s). Weights are block-shared through a
//   double-buffered LDS stage so L2 weight traffic is ~1GB instead of 8GB.
// ---------------------------------------------------------------------------

typedef __bf16 bf16;
typedef __attribute__((ext_vector_type(16))) __bf16 v16bf;
typedef __attribute__((ext_vector_type(8)))  __bf16 v8bf;
typedef __attribute__((ext_vector_type(4)))  __bf16 v4bf;
typedef __attribute__((ext_vector_type(8)))  float  v8f;
typedef __attribute__((ext_vector_type(4)))  float  v4f;

#define DIMX     512
#define CONDD    256
#define NHEAD    8
#define HDIM     64
#define BATCH    16
#define NSEQ     8192
#define NCOND    16
#define SCALEF   0.125f            // 64^-0.5

#define WAVES_PER_BLOCK 8
#define NTHREADS (WAVES_PER_BLOCK * 32)          // 256
#define ROWS_PER_BLOCK (WAVES_PER_BLOCK * 16)    // 128

// per-wave LDS: buf0 (x-tile then attn-out, 16x512 bf16) + qbuf (16x512 bf16)
//               + attn scratch (16x16 bf16)
#define LDS_PER_WAVE (16*512*2 + 16*512*2 + 16*16*2)          // 33280 B
#define WSTAGE_ELEMS (16 * DIMX)                              // one 16x512 strip
#define LDS_TOTAL (WAVES_PER_BLOCK*LDS_PER_WAVE + 2*WSTAGE_ELEMS*2)  // 292 KB

// wave-local LDS ordering fence (CDNA5 split counters; per-wave regions only)
__device__ __forceinline__ void lds_fence() {
  asm volatile("s_wait_dscnt 0" ::: "memory");
}

__device__ __forceinline__ v16bf cat8(v8bf lo, v8bf hi) {
  v16bf r;
#pragma unroll
  for (int i = 0; i < 8; ++i) { r[i] = lo[i]; r[8 + i] = hi[i]; }
  return r;
}

// A-matrix (16xK tile, row-major source, bf16). Contiguous K order (matches B):
// lanes 0-15: M=lane, K = k0..k0+15
// lanes 16-31: K = k0+16..k0+31.  -> two contiguous 8-elem (16B) loads.
__device__ __forceinline__ v16bf load_a16(const bf16* base, int stride, int k0,
                                          int lane) {
  const int m     = lane & 15;
  const int halfK = (lane >> 4) << 4;
  const bf16* p   = base + m * stride + k0 + halfK;
  v8bf lo = *(const v8bf*)(p);
  v8bf hi = *(const v8bf*)(p + 8);
  return cat8(lo, hi);
}

// B-matrix (Kx16 tile) taken from a row-major [outCol][K] bf16 array.
// Contiguous K order (matches A): lane n<16 holds col n, K=k0..k0+15;
// lanes 16-31 hold K=k0+16..k0+31.  -> one contiguous 32B load per lane.
__device__ __forceinline__ v16bf load_b16(const bf16* base, int stride, int k0,
                                          int lane) {
  const int n  = lane & 15;
  const int kb = k0 + ((lane >> 4) << 4);
  return *(const v16bf*)(base + n * stride + kb);
}

__device__ __forceinline__ v8f wmma_bf16(v16bf a, v16bf b, v8f c) {
  return __builtin_amdgcn_wmma_f32_16x16x32_bf16(false, a, false, b,
                                                 (short)0, c, false, false);
}

// 16KB weight strip staging: 1024 16B-chunks over 256 threads (4 each)
__device__ __forceinline__ void stage_load(const bf16* __restrict__ w, int nt,
                                           v8bf r[4], int tid) {
  const bf16* p = w + nt * 16 * DIMX;
#pragma unroll
  for (int i = 0; i < 4; ++i)
    r[i] = *(const v8bf*)(p + (tid + 256 * i) * 8);
}
// NOTE: 'ws' must stay an LDS-derived pointer with an integer offset select
// (never a pointer-select) so addrspace(3) inference survives -> ds_store.
__device__ __forceinline__ void stage_store(bf16* ws, const v8bf r[4],
                                            int tid) {
#pragma unroll
  for (int i = 0; i < 4; ++i)
    *(v8bf*)(ws + (tid + 256 * i) * 8) = r[i];
}

// ---------------------------------------------------------------------------
// Prep kernel 1: fp32 -> bf16 conversion of w_q and w_proj (row-major kept;
// row-major [out][in] is exactly the per-lane-contiguous B layout we need).
// ---------------------------------------------------------------------------
__global__ void __launch_bounds__(256)
mcca_cvt_weights(const float* __restrict__ wq, const float* __restrict__ wp,
                 bf16* __restrict__ wqb, bf16* __restrict__ wpb) {
  int i = blockIdx.x * 256 + threadIdx.x;
  if (i < DIMX * DIMX) {
    wqb[i] = (bf16)wq[i];
    wpb[i] = (bf16)wp[i];
  }
}

// ---------------------------------------------------------------------------
// Prep kernel 2: kv = conditions @ w_kv^T  (tiny: 256 rows x 1024 x 256)
//   k  -> kbf [b][h][c][d]  row-major (B operand for scores GEMM)
//   v  -> vtb [b][h][d][c]  (v transposed: B operand for attn@v GEMM)
// ---------------------------------------------------------------------------
__global__ void __launch_bounds__(256)
mcca_kv(const float* __restrict__ cond, const float* __restrict__ wkv,
        bf16* __restrict__ kbf, bf16* __restrict__ vtb) {
  __shared__ float crow[CONDD];
  const int b = blockIdx.x >> 4;
  const int c = blockIdx.x & 15;
  const float* cp = cond + (b * NCOND + c) * CONDD;
  if (threadIdx.x < CONDD) crow[threadIdx.x] = cp[threadIdx.x];
  __syncthreads();

  for (int o = threadIdx.x; o < 2 * DIMX; o += 256) {
    const float* w = wkv + o * CONDD;
    float acc = 0.f;
#pragma unroll 4
    for (int j = 0; j < CONDD; j += 4) {
      acc += crow[j] * w[j] + crow[j + 1] * w[j + 1] +
             crow[j + 2] * w[j + 2] + crow[j + 3] * w[j + 3];
    }
    if (o < DIMX) {                       // K part
      int h = o / HDIM, d = o % HDIM;
      kbf[((b * NHEAD + h) * NCOND + c) * HDIM + d] = (bf16)acc;
    } else {                              // V part, stored transposed
      int o2 = o - DIMX;
      int h = o2 / HDIM, d = o2 % HDIM;
      vtb[((b * NHEAD + h) * HDIM + d) * NCOND + c] = (bf16)acc;
    }
  }
}

// ---------------------------------------------------------------------------
// Fused main kernel. Per wave: one 16-row tile of x. Per block: 8 waves share
// each weight strip through a double-buffered 16KB LDS stage.
//   stage x->LDS bf16 ; q = x@wq^T (WMMA) -> LDS bf16 ;
//   per head: scores = q k^T (WMMA) -> mask -> softmax (shfl over 16 lanes)
//             -> attn (bf16, via LDS transpose) @ v^T (WMMA) -> LDS bf16 ;
//   out = attnout @ wproj^T + bias (WMMA) -> global f32.
// ---------------------------------------------------------------------------
__global__ void __launch_bounds__(NTHREADS)
mcca_fused(const float* __restrict__ x,
           const unsigned char* __restrict__ cmask,
           const bf16* __restrict__ wq, const bf16* __restrict__ wp,
           const bf16* __restrict__ kbf, const bf16* __restrict__ vtb,
           const float* __restrict__ bproj, float* __restrict__ out) {
  extern __shared__ char smem[];
  const int tid  = threadIdx.x;
  const int lane = tid & 31;
  const int wv   = tid >> 5;

  bf16* buf0   = (bf16*)(smem + wv * LDS_PER_WAVE);  // x tile, then attn-out
  bf16* qbuf   = buf0 + 16 * DIMX;                   // q tile (bf16)
  bf16* attn_s = qbuf + 16 * DIMX;                   // 16x16 attn transpose
  bf16* wstage = (bf16*)(smem + WAVES_PER_BLOCK * LDS_PER_WAVE); // 2 buffers

  const long row0 = (long)blockIdx.x * ROWS_PER_BLOCK + wv * 16;
  const int  b    = (int)(row0 / NSEQ);
  const int  ncol = lane & 15;
  const int  rbase = (lane >> 4) * 8;                // C/D row base per half

  // ---- stage x tile (16x512 f32 -> bf16 LDS, packed b64 stores) ------------
  const float* xrow = x + row0 * DIMX;
#pragma unroll 4
  for (int i = lane; i < 16 * DIMX / 4; i += 32) {
    v4f v = *(const v4f*)(xrow + i * 4);
    v4bf t;
    t[0] = (bf16)v[0]; t[1] = (bf16)v[1];
    t[2] = (bf16)v[2]; t[3] = (bf16)v[3];
    *(v4bf*)(buf0 + i * 4) = t;
  }
  lds_fence();

  // ---- q = x @ wq^T  (block-shared weight strips, double buffered) ---------
  {
    v8bf pre[4];
    stage_load(wq, 0, pre, tid);
    stage_store(wstage, pre, tid);
    __syncthreads();
    for (int nt = 0; nt < 32; ++nt) {
      const bf16* cur = wstage + (nt & 1) * WSTAGE_ELEMS;   // offset select!
      if (nt + 1 < 32) stage_load(wq, nt + 1, pre, tid);
      v8f acc = {};
#pragma unroll 4
      for (int kt = 0; kt < 16; ++kt) {
        v16bf a  = load_a16(buf0, DIMX, kt * 32, lane);
        v16bf bb = load_b16(cur, DIMX, kt * 32, lane);
        acc = wmma_bf16(a, bb, acc);
      }
      if (nt + 1 < 32)
        stage_store(wstage + ((nt + 1) & 1) * WSTAGE_ELEMS, pre, tid);
#pragma unroll
      for (int r = 0; r < 8; ++r)
        qbuf[(rbase + r) * DIMX + nt * 16 + ncol] = (bf16)acc[r];
      __syncthreads();   // next-buffer stores visible; frees buffer nt&1
    }
  }

  // ---- attention (C = 16 conditions, softmax across 16-lane halves) --------
  const bool valid = cmask[b * NCOND + ncol] != 0;  // lane's condition col

  for (int h = 0; h < NHEAD; ++h) {
    const bf16* kp = kbf + (b * NHEAD + h) * (NCOND * HDIM);
    v8f s = {};
#pragma unroll
    for (int kt = 0; kt < 2; ++kt) {
      v16bf a  = load_a16(qbuf + h * HDIM, DIMX, kt * 32, lane);
      v16bf bb = load_b16(kp, HDIM, kt * 32, lane);
      s = wmma_bf16(a, bb, s);
    }

    // mask + softmax over c (each VGPR r / lane-half is one row)
    v8f p;
#pragma unroll
    for (int r = 0; r < 8; ++r) {
      float sc = valid ? s[r] * SCALEF : -__builtin_inff();
      float mx = sc;
#pragma unroll
      for (int off = 8; off >= 1; off >>= 1)
        mx = fmaxf(mx, __shfl_xor(mx, off, 16));
      float e = __expf(sc - mx);         // mask[:,0]==True -> mx finite
      float sum = e;
#pragma unroll
      for (int off = 8; off >= 1; off >>= 1)
        sum += __shfl_xor(sum, off, 16);
      p[r] = e / sum;
    }

    // D-layout -> A-layout via 16x16 LDS scratch (bf16)
#pragma unroll
    for (int r = 0; r < 8; ++r)
      attn_s[(rbase + r) * 16 + ncol] = (bf16)p[r];
    lds_fence();

    v16bf a;
    {
      const bool lowh = lane < 16;                 // lanes 0-15: K = 0..15, lanes 16-31 pad
      v8bf lo = *(const v8bf*)(attn_s + ncol * 16);
      v8bf hi = *(const v8bf*)(attn_s + ncol * 16 + 8);
#pragma unroll
      for (int i = 0; i < 8; ++i) { a[i] = lowh ? lo[i] : (bf16)0.f; a[8 + i] = lowh ? hi[i] : (bf16)0.f; }
    }

    const bf16* vp = vtb + (b * NHEAD + h) * (HDIM * NCOND);
#pragma unroll
    for (int nt = 0; nt < 4; ++nt) {
      v16bf bb;
      if (lane < 16) {                   // K rows 0..15 (c), lanes 16-31 pad
        bb = *(const v16bf*)(vp + (nt * 16 + ncol) * NCOND);
      } else {
#pragma unroll
        for (int i = 0; i < 16; ++i) bb[i] = (bf16)0.f;
      }
      v8f z = {};
      v8f o = wmma_bf16(a, bb, z);
#pragma unroll
      for (int r = 0; r < 8; ++r)
        buf0[(rbase + r) * DIMX + h * HDIM + nt * 16 + ncol] = (bf16)o[r];
    }
    lds_fence();                          // scratch reused next head
  }

  // ---- out = attnout @ wproj^T + bias (block-shared strips again) ----------
  {
    v8bf pre[4];
    stage_load(wp, 0, pre, tid);
    __syncthreads();                      // all waves done with attention LDS
    stage_store(wstage, pre, tid);
    __syncthreads();
    for (int nt = 0; nt < 32; ++nt) {
      const bf16* cur = wstage + (nt & 1) * WSTAGE_ELEMS;   // offset select!
      if (nt + 1 < 32) stage_load(wp, nt + 1, pre, tid);
      v8f acc = {};
#pragma unroll 4
      for (int kt = 0; kt < 16; ++kt) {
        v16bf a  = load_a16(buf0, DIMX, kt * 32, lane);
        v16bf bb = load_b16(cur, DIMX, kt * 32, lane);
        acc = wmma_bf16(a, bb, acc);
      }
      if (nt + 1 < 32)
        stage_store(wstage + ((nt + 1) & 1) * WSTAGE_ELEMS, pre, tid);
      const float bias = bproj[nt * 16 + ncol];
      float* op = out + row0 * DIMX + nt * 16 + ncol;
#pragma unroll
      for (int r = 0; r < 8; ++r)
        op[(rbase + r) * DIMX] = acc[r] + bias;
      __syncthreads();
    }
  }
}

// ---------------------------------------------------------------------------
extern "C" void kernel_launch(void* const* d_in, const int* in_sizes, int n_in,
                              void* d_out, int out_size, void* d_ws,
                              size_t ws_size, hipStream_t stream) {
  const float*         x      = (const float*)d_in[0];
  const float*         cond   = (const float*)d_in[1];
  const unsigned char* cmask  = (const unsigned char*)d_in[2];  // bool mask
  const float*         w_q    = (const float*)d_in[3];
  const float*         w_kv   = (const float*)d_in[4];
  const float*         w_proj = (const float*)d_in[5];
  const float*         b_proj = (const float*)d_in[6];
  float*               out    = (float*)d_out;

  char* ws = (char*)d_ws;
  bf16* wqb = (bf16*)ws;  ws += (size_t)DIMX * DIMX * 2;
  bf16* wpb = (bf16*)ws;  ws += (size_t)DIMX * DIMX * 2;
  bf16* kbf = (bf16*)ws;  ws += (size_t)BATCH * NHEAD * NCOND * HDIM * 2;
  bf16* vtb = (bf16*)ws;  ws += (size_t)BATCH * NHEAD * NCOND * HDIM * 2;

  mcca_cvt_weights<<<(DIMX * DIMX + 255) / 256, 256, 0, stream>>>(w_q, w_proj,
                                                                  wqb, wpb);
  mcca_kv<<<BATCH * NCOND, 256, 0, stream>>>(cond, w_kv, kbf, vtb);

  const size_t shmem = LDS_TOTAL;                          // ~292 KB
  dim3 grid((BATCH * NSEQ) / ROWS_PER_BLOCK);              // 1024 blocks
  mcca_fused<<<grid, NTHREADS, shmem, stream>>>(
      x, cmask, wqb, wpb, kbf, vtb, b_proj, out);
}
